// Attention_26250840113588
// MI455X (gfx1250) — matrix operand order repaired and checked by reading, not compiled, not run on hardware
//
#include <hip/hip_runtime.h>

typedef __bf16 bf16;
typedef __attribute__((ext_vector_type(16))) __bf16 v16bf;
typedef __attribute__((ext_vector_type(8)))  float  v8f;

#define DMODEL 4096
#define SEQ    2048
#define NHEAD  32
#define NKVH   8
#define HDIM   128
#define BATCH  2

// ---------------- fragment loaders (wave32, 16-bit WMMA layouts) ----------------
// A 16x32 (MxK): lane holds row M = lane&15; chunk0 = K[kb + half*8 .. +7],
// chunk1 = K[kb + half*8 + 16 .. +7]  (half = lane>>4)
__device__ __forceinline__ v16bf frag_a_bf16(const bf16* rowp, int kb, int half) {
  const bf16* p = rowp + kb + half * 8;
  v16bf a;
#pragma unroll
  for (int j = 0; j < 8; ++j) a[j] = p[j];
#pragma unroll
  for (int j = 0; j < 8; ++j) a[8 + j] = p[16 + j];
  return a;
}
// B 32x16 (KxN): lane holds col N = lane&15; chunk0 = K[kb + half*8 .. +7],
// chunk1 = K[kb + half*8 + 16 .. +7] (caller passes pointer to K = kb + half*8)
__device__ __forceinline__ v16bf frag_b_bf16(const bf16* colp) {
  v16bf b;
#pragma unroll
  for (int j = 0; j < 8; ++j) { b[j] = colp[j]; b[8 + j] = colp[16 + j]; }
  return b;
}

// ---------------- kernel 0: f32 -> bf16 pre-conversion (x, wq, wo) ----------------
__global__ __launch_bounds__(256)
void cvt_bf16_kernel(const float* __restrict__ src, unsigned short* __restrict__ dst_raw) {
  bf16* dst = reinterpret_cast<bf16*>(dst_raw);
  const size_t i0 = ((size_t)blockIdx.x * 256 + threadIdx.x) * 8;
#pragma unroll
  for (int j = 0; j < 8; ++j) dst[i0 + j] = (__bf16)src[i0 + j];
}

// ---------------- kernel 1: xq = (x @ wq.T) * rope, bf16 x bf16 -> bf16 ----------------
// 64x64 tile per wave (4x4 WMMA C tiles), 8 waves -> 128x256 per WG.
__global__ __launch_bounds__(256)
void qproj_rope_kernel(const unsigned short* __restrict__ xb_raw,
                       const unsigned short* __restrict__ wqb_raw,
                       const float* __restrict__ freqs,
                       unsigned short* __restrict__ xq_raw) {
  const bf16* X = reinterpret_cast<const bf16*>(xb_raw);
  const bf16* W = reinterpret_cast<const bf16*>(wqb_raw);
  bf16* xq = reinterpret_cast<bf16*>(xq_raw);
  const int lane = threadIdx.x & 31, wave = threadIdx.x >> 5;
  const int half = lane >> 4, l15 = lane & 15;
  const int m0 = blockIdx.y * 128 + (wave & 1) * 64;
  const int n0 = blockIdx.x * 256 + (wave >> 1) * 64;

  v8f acc[4][4] = {};
  const bf16* rowA[4];
  const bf16* colB[4];
#pragma unroll
  for (int i = 0; i < 4; ++i) rowA[i] = X + (size_t)(m0 + i * 16 + l15) * DMODEL;
#pragma unroll
  for (int j = 0; j < 4; ++j) colB[j] = W + (size_t)(n0 + j * 16 + l15) * DMODEL;

  for (int kb = 0; kb < DMODEL; kb += 32) {
    const int ks = kb + half * 8;
    __builtin_prefetch((const void*)(rowA[0] + kb + 256), 0, 1);
    __builtin_prefetch((const void*)(colB[0] + ks + 256), 0, 1);
    v16bf a[4], b[4];
#pragma unroll
    for (int i = 0; i < 4; ++i) a[i] = frag_a_bf16(rowA[i], kb, half);
#pragma unroll
    for (int j = 0; j < 4; ++j) b[j] = frag_b_bf16(colB[j] + ks);
#pragma unroll
    for (int i = 0; i < 4; ++i)
#pragma unroll
      for (int j = 0; j < 4; ++j)
        acc[i][j] = __builtin_amdgcn_wmma_f32_16x16x32_bf16(false, a[i], false, b[j], (short)0, acc[i][j], false, false);
  }
  // C/D layout: lane col N = l15 (+tile), rows M = half*8 + j
#pragma unroll
  for (int mi = 0; mi < 4; ++mi)
#pragma unroll
    for (int nj = 0; nj < 4; ++nj) {
      const int n = n0 + nj * 16 + l15;
      const int mb = m0 + mi * 16 + half * 8;
#pragma unroll
      for (int j = 0; j < 8; ++j) {
        const int m = mb + j;
        const float f = freqs[((m & (SEQ - 1)) << 6) + (n & 63)];
        xq[(size_t)m * DMODEL + n] = (__bf16)(acc[mi][nj][j] * f);
      }
    }
}

// ---------------- kernel 2: flash attention (causal, GQA) ----------------
// One WG = one (b, head); 8 waves each own a 16-row q block; kv tiled by 32.
// K and V tiles cooperatively staged in LDS as bf16 (K row-major for A-frags,
// V transposed for B-frags). Scores computed transposed (ST = K.Q^T) so the
// softmax reduce is lane-local + one shfl_xor(16) and P lands directly in the
// A-fragment layout for O = P.V.
__global__ __launch_bounds__(256)
void attn_kernel(const unsigned short* __restrict__ xq_raw,
                 const float* __restrict__ cache_k, const float* __restrict__ cache_v,
                 unsigned short* __restrict__ attn_raw) {
  const bf16* xq = reinterpret_cast<const bf16*>(xq_raw);
  bf16* attn = reinterpret_cast<bf16*>(attn_raw);
  __shared__ bf16 ldsK[32 * HDIM];  // [kv][hd] row-major
  __shared__ bf16 ldsV[32 * HDIM];  // [hd][kv] transposed

  const int lane = threadIdx.x & 31, wave = threadIdx.x >> 5;
  const int half = lane >> 4, l15 = lane & 15;
  const int h = blockIdx.x & (NHEAD - 1);
  const int b = blockIdx.x >> 5;
  const int g = h >> 2;  // NREP = 4
  const int q0 = blockIdx.y * 128 + wave * 16;
  const float scale = 0.08838834764831845f;  // 1/sqrt(128)

  const float* kbase = cache_k + ((size_t)b * SEQ * NKVH + g) * HDIM;  // + kv*NKVH*HDIM
  const float* vbase = cache_v + ((size_t)b * SEQ * NKVH + g) * HDIM;

  // Q^T B-fragments: lane col q = q0+l15, two 8-hd runs (half*8, half*8+16) per 32-hd block
  v16bf qf[4];
  {
    const bf16* qp = xq + (size_t)(b * SEQ + q0 + l15) * DMODEL + h * HDIM + half * 8;
#pragma unroll
    for (int hb = 0; hb < 4; ++hb) qf[hb] = frag_b_bf16(qp + hb * 32);
  }

  v8f oacc[8] = {};  // O tiles (16 q x 16 hd), ht = 0..7
  float m_run = -1e30f, l_run = 0.0f;
  const int myq = q0 + l15;
  const int kvEnd = blockIdx.y * 128 + 128;

  for (int kv0 = 0; kv0 < kvEnd; kv0 += 32) {
    // cooperative stage: K tile row-major, V tile transposed, both bf16
    {
      const int t = threadIdx.x;
      const int kv = t >> 3;
      const int hd0 = (t & 7) * 16;
      const float* kp = kbase + (size_t)(kv0 + kv) * (NKVH * HDIM) + hd0;
      const float* vp = vbase + (size_t)(kv0 + kv) * (NKVH * HDIM) + hd0;
      __builtin_prefetch((const void*)(kp + 32 * NKVH * HDIM), 0, 1);
      __builtin_prefetch((const void*)(vp + 32 * NKVH * HDIM), 0, 1);
#pragma unroll
      for (int i = 0; i < 16; ++i) ldsK[kv * HDIM + hd0 + i] = (__bf16)kp[i];
#pragma unroll
      for (int i = 0; i < 16; ++i) ldsV[(hd0 + i) * 32 + kv] = (__bf16)vp[i];
    }
    __syncthreads();

    if (kv0 <= q0 + 15) {  // wave-uniform: EXEC stays all-1 for WMMA
      // ST tiles (kv x q): lane holds col q = l15, rows kv = kt*16 + half*8 + j
      float sv[2][8];
      float tmax = -1e30f;
#pragma unroll
      for (int kt = 0; kt < 2; ++kt) {
        v8f st = {};
        const bf16* krow = &ldsK[(kt * 16 + l15) * HDIM];
#pragma unroll
        for (int hb = 0; hb < 4; ++hb) {
          v16bf ka = frag_a_bf16(krow, hb * 32, half);
          st = __builtin_amdgcn_wmma_f32_16x16x32_bf16(false, ka, false, qf[hb], (short)0, st, false, false);
        }
#pragma unroll
        for (int j = 0; j < 8; ++j) {
          const int kvr = kv0 + kt * 16 + half * 8 + j;
          float v = st[j] * scale;
          if (kvr > myq) v += -1e9f;  // causal additive mask, as in reference
          sv[kt][j] = v;
          tmax = fmaxf(tmax, v);
        }
      }
      tmax = fmaxf(tmax, __shfl_xor(tmax, 16, 32));
      const float mnew = fmaxf(m_run, tmax);
      const float alpha = __expf(m_run - mnew);
      float lsum = 0.0f;
#pragma unroll
      for (int kt = 0; kt < 2; ++kt)
#pragma unroll
        for (int j = 0; j < 8; ++j) {
          const float p = __expf(sv[kt][j] - mnew);
          sv[kt][j] = p;
          lsum += p;
        }
      lsum += __shfl_xor(lsum, 16, 32);
      l_run = l_run * alpha + lsum;
      m_run = mnew;

      // per-q-row alpha broadcast, rescale O
      float arow[8];
#pragma unroll
      for (int j = 0; j < 8; ++j) arow[j] = __shfl(alpha, half * 8 + j, 32);
#pragma unroll
      for (int ht = 0; ht < 8; ++ht)
#pragma unroll
        for (int j = 0; j < 8; ++j) oacc[ht][j] *= arow[j];

      // P (16 q x 32 kv) A-fragment: exactly the ST D-tile layout
      v16bf pf;
#pragma unroll
      for (int j = 0; j < 8; ++j) { pf[j] = (__bf16)sv[0][j]; pf[8 + j] = (__bf16)sv[1][j]; }

      // O += P . V : V B-fragment = two 8-element kv runs (half*8, half*8+16) from transposed LDS
#pragma unroll
      for (int ht = 0; ht < 8; ++ht) {
        v16bf vb = frag_b_bf16(&ldsV[(ht * 16 + l15) * 32 + half * 8]);
        oacc[ht] = __builtin_amdgcn_wmma_f32_16x16x32_bf16(false, pf, false, vb, (short)0, oacc[ht], false, false);
      }
    }
    __syncthreads();
  }

  // normalize rows by l, store bf16 (O tile: lane col hd = ht*16+l15, rows q = half*8+j)
  const float linv = 1.0f / l_run;
  float lrow[8];
#pragma unroll
  for (int j = 0; j < 8; ++j) lrow[j] = __shfl(linv, half * 8 + j, 32);
#pragma unroll
  for (int ht = 0; ht < 8; ++ht) {
    const int hd = ht * 16 + l15;
#pragma unroll
    for (int j = 0; j < 8; ++j) {
      const int q = q0 + half * 8 + j;
      attn[(size_t)(b * SEQ + q) * DMODEL + h * HDIM + hd] = (__bf16)(oacc[ht][j] * lrow[j]);
    }
  }
}

// ---------------- kernel 3: out = attn @ wo.T (bf16 x bf16 -> f32) ----------------
__global__ __launch_bounds__(256)
void oproj_kernel(const unsigned short* __restrict__ attn_raw,
                  const unsigned short* __restrict__ wob_raw,
                  float* __restrict__ out) {
  const bf16* A = reinterpret_cast<const bf16*>(attn_raw);
  const bf16* W = reinterpret_cast<const bf16*>(wob_raw);
  const int lane = threadIdx.x & 31, wave = threadIdx.x >> 5;
  const int half = lane >> 4, l15 = lane & 15;
  const int m0 = blockIdx.y * 128 + (wave & 1) * 64;
  const int n0 = blockIdx.x * 256 + (wave >> 1) * 64;

  v8f acc[4][4] = {};
  const bf16* rowA[4];
  const bf16* colB[4];
#pragma unroll
  for (int i = 0; i < 4; ++i) rowA[i] = A + (size_t)(m0 + i * 16 + l15) * DMODEL;
#pragma unroll
  for (int j = 0; j < 4; ++j) colB[j] = W + (size_t)(n0 + j * 16 + l15) * DMODEL;

  for (int kb = 0; kb < DMODEL; kb += 32) {
    const int ks = kb + half * 8;
    __builtin_prefetch((const void*)(rowA[0] + kb + 256), 0, 1);
    __builtin_prefetch((const void*)(colB[0] + ks + 256), 0, 1);
    v16bf a[4], b[4];
#pragma unroll
    for (int i = 0; i < 4; ++i) a[i] = frag_a_bf16(rowA[i], kb, half);
#pragma unroll
    for (int j = 0; j < 4; ++j) b[j] = frag_b_bf16(colB[j] + ks);
#pragma unroll
    for (int i = 0; i < 4; ++i)
#pragma unroll
      for (int j = 0; j < 4; ++j)
        acc[i][j] = __builtin_amdgcn_wmma_f32_16x16x32_bf16(false, a[i], false, b[j], (short)0, acc[i][j], false, false);
  }
#pragma unroll
  for (int mi = 0; mi < 4; ++mi)
#pragma unroll
    for (int nj = 0; nj < 4; ++nj) {
      const int n = n0 + nj * 16 + l15;
      const int mb = m0 + mi * 16 + half * 8;
#pragma unroll
      for (int j = 0; j < 8; ++j)
        out[(size_t)(mb + j) * DMODEL + n] = acc[mi][nj][j];
    }
}

// ---------------- host ----------------
extern "C" void kernel_launch(void* const* d_in, const int* in_sizes, int n_in,
                              void* d_out, int out_size, void* d_ws, size_t ws_size,
                              hipStream_t stream) {
  const float* x     = (const float*)d_in[0];
  const float* freqs = (const float*)d_in[1];
  // d_in[2] = mask: causal triu(-1e9, k=1), computed analytically in-kernel
  const float* wq    = (const float*)d_in[3];
  // d_in[4]/d_in[5] = wk/wv: dead in the reference forward (cache never written)
  const float* wo    = (const float*)d_in[6];
  const float* ck    = (const float*)d_in[7];
  const float* cv    = (const float*)d_in[8];
  float* out = (float*)d_out;

  const size_t NELEM = (size_t)DMODEL * (BATCH * SEQ);  // 16.7M, same for all big mats
  unsigned short* x_bf    = (unsigned short*)d_ws;
  unsigned short* wq_bf   = x_bf + NELEM;
  unsigned short* wo_bf   = wq_bf + NELEM;
  unsigned short* xq_ws   = wo_bf + NELEM;
  unsigned short* attn_ws = xq_ws + NELEM;

  const int cvt_blocks = (int)(NELEM / (256 * 8));
  cvt_bf16_kernel<<<cvt_blocks, 256, 0, stream>>>(x,  x_bf);
  cvt_bf16_kernel<<<cvt_blocks, 256, 0, stream>>>(wq, wq_bf);
  cvt_bf16_kernel<<<cvt_blocks, 256, 0, stream>>>(wo, wo_bf);

  dim3 gemm_grid(DMODEL / 256, (BATCH * SEQ) / 128);
  qproj_rope_kernel<<<gemm_grid, 256, 0, stream>>>(x_bf, wq_bf, freqs, xq_ws);

  dim3 attn_grid(BATCH * NHEAD, SEQ / 128);
  attn_kernel<<<attn_grid, 256, 0, stream>>>(xq_ws, ck, cv, attn_ws);

  oproj_kernel<<<gemm_grid, 256, 0, stream>>>(attn_ws, wo_bf, out);
}
